// PNetDecoder_70892730188269
// MI455X (gfx1250) — matrix operand order repaired and checked by reading, not compiled, not run on hardware
//
#include <hip/hip_runtime.h>
#include <math.h>

#define BATCH   128
#define NNODE   500
#define NM      (BATCH*NNODE)     // 64000 nodes
#define NEDGE   1048576
#define D       128
#define F_IN    5
#define EDGE_DIM 2
#define VN      10
#define NEG_FEAT 0.01f
#define NEG_ATT  0.2f

typedef __attribute__((ext_vector_type(16))) __bf16 v16bf;
typedef __attribute__((ext_vector_type(8)))  float  v8f;

__device__ __forceinline__ float leaky(float x, float s) {
    return fmaxf(x, 0.0f) + s * fminf(x, 0.0f);
}

// round-to-nearest-even f32 -> bf16 (as raw u16)
__device__ __forceinline__ unsigned short f2bf(float f) {
    unsigned u = __float_as_uint(f);
    u += 0x7FFFu + ((u >> 16) & 1u);
    return (unsigned short)(u >> 16);
}

// order-preserving float -> uint key (monotone for all finite floats)
__device__ __forceinline__ unsigned fkey(float f) {
    unsigned u = __float_as_uint(f);
    unsigned m = (unsigned)(((int)u) >> 31) | 0x80000000u;
    return u ^ m;
}
__device__ __forceinline__ float keyf(unsigned k) {
    unsigned u = (k & 0x80000000u) ? (k ^ 0x80000000u) : ~k;
    return __uint_as_float(u);
}

// ---------------------------------------------------------------- utilities
__global__ void k_zero(unsigned* p, long n) {
    long i = (long)blockIdx.x * blockDim.x + threadIdx.x;
    if (i < n) p[i] = 0u;
}

// h0 = leaky(x@pW + pb, .01) + vge[b] + vnd[b, curr_id[b]]   (block = node)
// also emits the bf16 copy for the first WMMA GEMM.
__global__ void k_input_proj(const float* __restrict__ x, const float* __restrict__ pW,
                             const float* __restrict__ pb, const int* __restrict__ cvid,
                             const float* __restrict__ vge, const float* __restrict__ vnd,
                             float* __restrict__ h, unsigned short* __restrict__ hb) {
    int n = blockIdx.x;
    int d = threadIdx.x;
    int b = n / NNODE;
    float acc = pb[d];
#pragma unroll
    for (int k = 0; k < F_IN; ++k) acc += x[n * F_IN + k] * pW[k * D + d];
    acc = leaky(acc, NEG_FEAT);
    int id = cvid[b];
    acc += vge[b * D + d] + vnd[(b * VN + id) * D + d];
    long i = (long)n * D + d;
    h[i]  = acc;
    hb[i] = f2bf(acc);
}

// Wt[n,k] = bf16(W[k,n])  (so B operand loads contiguous-K dwords)
__global__ void k_w_transpose(const float* __restrict__ W, unsigned short* __restrict__ Wt) {
    int i = blockIdx.x * blockDim.x + threadIdx.x;   // 0 .. D*D-1
    int k = i / D, n = i % D;
    Wt[n * D + k] = f2bf(W[k * D + n]);
}

// ce[r] = sum_d We[r,d] * a_edge[d]   (tiny)
__global__ void k_ce(const float* __restrict__ We, const float* __restrict__ a_edge,
                     float* __restrict__ ce) {
    int r = threadIdx.x;
    if (r < EDGE_DIM) {
        float s = 0.0f;
        for (int d = 0; d < D; ++d) s += We[r * D + d] * a_edge[d];
        ce[r] = s;
    }
}

// ------------------------------------------------------- WMMA GEMM  [NM,128]@[128,128]
// block = 256 threads = 8 waves; blockIdx.x = group of 4 M tiles (64 rows);
// wave = N tile (16 cols). B (Wt, 32KB) is staged into LDS with CDNA5 async
// copies (global_load_async_to_lds_b128, ASYNCcnt), then each wave keeps all
// four K-step B operands in registers and streams A through 16 WMMAs.
__global__ void __launch_bounds__(256) k_gemm_bf16(const unsigned short* __restrict__ A,
                                                   const unsigned short* __restrict__ Bt,
                                                   float* __restrict__ C) {
    __shared__ unsigned short lwt[D * D];            // 32 KB

    // ---- async stage Bt -> LDS: 256 threads x 16B x 8 passes
    {
        int t = threadIdx.x;
#pragma unroll
        for (int it = 0; it < 8; ++it) {
            unsigned loff = (unsigned)(size_t)(&lwt[t * 8 + it * 2048]);   // LDS byte addr
            unsigned long long gaddr =
                (unsigned long long)(size_t)(Bt + t * 8 + it * 2048);      // global addr
            asm volatile("global_load_async_to_lds_b128 %0, %1, off"
                         :: "v"(loff), "v"(gaddr) : "memory");
        }
    }
    asm volatile("s_wait_asynccnt 0x0" ::: "memory");
    __syncthreads();

    int lane = threadIdx.x & 31;
    int wave = threadIdx.x >> 5;          // N tile 0..7
    int n0 = wave * 16;
    int hi = (lane >> 4) & 1;
    int bcol = n0 + (lane & 15);

    // ---- load all four K-step B operands from LDS into registers
    union BU { v16bf v; unsigned u[8]; };
    BU bk[4];
#pragma unroll
    for (int kbi = 0; kbi < 4; ++kbi) {
        int kbB = kbi * 32 + hi * 8;      // B: lanes 0-15 K=+0..7/+16..23, lanes 16-31 +8
#pragma unroll
        for (int j = 0; j < 8; ++j)
            bk[kbi].u[j] = *(const unsigned*)(&lwt[bcol * D + kbB + 2 * j + (j >= 4 ? 8 : 0)]);
    }

    // ---- stream 4 M tiles
    int m0base = blockIdx.x * 64;
#pragma unroll
    for (int mt = 0; mt < 4; ++mt) {
        int m0 = m0base + mt * 16;
        const unsigned short* ap = A + (long)(m0 + (lane & 15)) * D;
        v8f acc = {};
#pragma unroll
        for (int kbi = 0; kbi < 4; ++kbi) {
            union BU a;
            int ka = kbi * 32 + hi * 8;   // A: lanes 0-15 K=+0..7/+16..23, lanes 16-31 +8
#pragma unroll
            for (int j = 0; j < 4; ++j) {
                a.u[j]     = *(const unsigned*)(ap + ka + 2 * j);
                a.u[4 + j] = *(const unsigned*)(ap + ka + 16 + 2 * j);
            }
            acc = __builtin_amdgcn_wmma_f32_16x16x32_bf16(false, a.v, false, bk[kbi].v,
                                                          (short)0, acc, false, false);
        }
        // C layout: VGPR r -> row m0 + r + hi*8, col n0 + (lane&15)
        float* cp = C + (long)(m0 + hi * 8) * D + bcol;
#pragma unroll
        for (int r = 0; r < 8; ++r) cp[(long)r * D] = acc[r];
    }
}

// per-node attention scores: ssrc[n] = hW[n]·a_src, sdst[n] = hW[n]·a_dst  (wave per node)
__global__ void k_node_scores(const float* __restrict__ hW, const float* __restrict__ a_src,
                              const float* __restrict__ a_dst, float* __restrict__ ssrc,
                              float* __restrict__ sdst) {
    int lane = threadIdx.x & 31;
    int node = blockIdx.x * 8 + (threadIdx.x >> 5);
    float4 h4 = ((const float4*)(hW + (long)node * D))[lane];
    float4 s4 = ((const float4*)a_src)[lane];
    float4 d4 = ((const float4*)a_dst)[lane];
    float ps = h4.x * s4.x + h4.y * s4.y + h4.z * s4.z + h4.w * s4.w;
    float pd = h4.x * d4.x + h4.y * d4.y + h4.z * d4.z + h4.w * d4.w;
    for (int off = 16; off; off >>= 1) {
        ps += __shfl_down(ps, off, 32);
        pd += __shfl_down(pd, off, 32);
    }
    if (lane == 0) { ssrc[node] = ps; sdst[node] = pd; }
}

// pass 1: logits + segment max via ordered-key atomicMax
__global__ void k_edge_logits(const int* __restrict__ ei, const float* __restrict__ ea,
                              const float* __restrict__ ssrc, const float* __restrict__ sdst,
                              const float* __restrict__ ce, float* __restrict__ logits,
                              unsigned* __restrict__ mkey) {
    int e = blockIdx.x * blockDim.x + threadIdx.x;
    if (e >= NEDGE) return;
    int s = ei[e], d = ei[NEDGE + e];
    float lg = leaky(ssrc[s] + sdst[d] + ea[2 * e] * ce[0] + ea[2 * e + 1] * ce[1], NEG_ATT);
    logits[e] = lg;
    atomicMax(mkey + d, fkey(lg));
}

// pass 2: exp(logit - max) + segment sum
__global__ void k_edge_exp(const int* __restrict__ ei, const float* __restrict__ logits,
                           const unsigned* __restrict__ mkey, float* __restrict__ ex,
                           float* __restrict__ denom) {
    int e = blockIdx.x * blockDim.x + threadIdx.x;
    if (e >= NEDGE) return;
    int d = ei[NEDGE + e];
    float v = __expf(logits[e] - keyf(mkey[d]));
    ex[e] = v;
    atomicAdd(denom + d, v);
}

// pass 3: out[dst] += alpha * (hW[src] + edge_attr@We)   (wave per edge, float4 lanes)
__global__ void __launch_bounds__(256) k_edge_aggregate(const int* __restrict__ ei,
        const float* __restrict__ ea, const float* __restrict__ hW,
        const float* __restrict__ We, const float* __restrict__ ex,
        const float* __restrict__ denom, float* __restrict__ out) {
    int lane = threadIdx.x & 31;
    int e = blockIdx.x * 8 + (threadIdx.x >> 5);
    int s = ei[e], d = ei[NEDGE + e];
    float alpha = ex[e] / (denom[d] + 1e-16f);
    float ea0 = ea[2 * e], ea1 = ea[2 * e + 1];
    float4 h4 = ((const float4*)(hW + (long)s * D))[lane];
    float4 w0 = ((const float4*)We)[lane];
    float4 w1 = ((const float4*)(We + D))[lane];
    float* op = out + (long)d * D + lane * 4;
    atomicAdd(op + 0, alpha * (h4.x + ea0 * w0.x + ea1 * w1.x));
    atomicAdd(op + 1, alpha * (h4.y + ea0 * w0.y + ea1 * w1.y));
    atomicAdd(op + 2, alpha * (h4.z + ea0 * w0.z + ea1 * w1.z));
    atomicAdd(op + 3, alpha * (h4.w + ea0 * w0.w + ea1 * w1.w));
}

// h = leaky(acc + b, .01), also emit bf16 copy for the next layer's GEMM
__global__ void k_bias_act(const float* __restrict__ acc, const float* __restrict__ b,
                           float* __restrict__ h, unsigned short* __restrict__ hb, long n) {
    long i = (long)blockIdx.x * blockDim.x + threadIdx.x;
    if (i < n) {
        float v = leaky(acc[i] + b[i & (D - 1)], NEG_FEAT);
        h[i]  = v;
        hb[i] = f2bf(v);
    }
}

// ---------------------------------------------------------------- final D->1 layer
__global__ void k_final_node(const float* __restrict__ h, const float* __restrict__ W5,
                             float* __restrict__ hw5) {
    int lane = threadIdx.x & 31;
    int node = blockIdx.x * 8 + (threadIdx.x >> 5);
    float4 h4 = ((const float4*)(h + (long)node * D))[lane];
    float4 w4 = ((const float4*)W5)[lane];
    float ps = h4.x * w4.x + h4.y * w4.y + h4.z * w4.z + h4.w * w4.w;
    for (int off = 16; off; off >>= 1) ps += __shfl_down(ps, off, 32);
    if (lane == 0) hw5[node] = ps;
}

__global__ void k_edge_logits5(const int* __restrict__ ei, const float* __restrict__ ea,
                               const float* __restrict__ hw5, const float* __restrict__ a_src5,
                               const float* __restrict__ a_dst5, const float* __restrict__ a_edge5,
                               const float* __restrict__ We5, float* __restrict__ logits,
                               unsigned* __restrict__ mkey) {
    int e = blockIdx.x * blockDim.x + threadIdx.x;
    if (e >= NEDGE) return;
    int s = ei[e], d = ei[NEDGE + e];
    float e5 = ea[2 * e] * We5[0] + ea[2 * e + 1] * We5[1];
    float lg = leaky(a_src5[0] * hw5[s] + a_dst5[0] * hw5[d] + a_edge5[0] * e5, NEG_ATT);
    logits[e] = lg;
    atomicMax(mkey + d, fkey(lg));
}

__global__ void k_edge_agg5(const int* __restrict__ ei, const float* __restrict__ ea,
                            const float* __restrict__ hw5, const float* __restrict__ We5,
                            const float* __restrict__ ex, const float* __restrict__ denom,
                            float* __restrict__ outf) {
    int e = blockIdx.x * blockDim.x + threadIdx.x;
    if (e >= NEDGE) return;
    int s = ei[e], d = ei[NEDGE + e];
    float alpha = ex[e] / (denom[d] + 1e-16f);
    float e5 = ea[2 * e] * We5[0] + ea[2 * e + 1] * We5[1];
    atomicAdd(outf + d, alpha * (hw5[s] + e5));
}

__global__ void k_final_out(const float* __restrict__ outf, const float* __restrict__ b5,
                            float* __restrict__ y) {
    int i = blockIdx.x * blockDim.x + threadIdx.x;
    if (i < NM) y[i] = outf[i] + b5[0];
}

// ---------------------------------------------------------------- launch
extern "C" void kernel_launch(void* const* d_in, const int* in_sizes, int n_in,
                              void* d_out, int out_size, void* d_ws, size_t ws_size,
                              hipStream_t stream) {
    (void)in_sizes; (void)n_in; (void)out_size; (void)ws_size;
    const float* x      = (const float*)d_in[0];
    const int*   ei     = (const int*)d_in[1];
    const float* ea     = (const float*)d_in[2];
    const int*   cvid   = (const int*)d_in[3];
    const float* vge    = (const float*)d_in[4];
    const float* vnd    = (const float*)d_in[5];
    const float* pW     = (const float*)d_in[6];
    const float* pb     = (const float*)d_in[7];
    const float* Ws     = (const float*)d_in[8];
    const float* bs     = (const float*)d_in[9];
    const float* a_srcs = (const float*)d_in[10];
    const float* a_dsts = (const float*)d_in[11];
    const float* a_edges= (const float*)d_in[12];
    const float* Wes    = (const float*)d_in[13];
    const float* W5     = (const float*)d_in[14];
    const float* b5     = (const float*)d_in[15];
    const float* a_src5 = (const float*)d_in[16];
    const float* a_dst5 = (const float*)d_in[17];
    const float* a_edge5= (const float*)d_in[18];
    const float* We5    = (const float*)d_in[19];
    float* y = (float*)d_out;

    char* p = (char*)d_ws;
    auto carve = [&](size_t bytes) -> char* {
        char* r = p;
        p += (bytes + 255) & ~(size_t)255;
        return r;
    };
    float*          h     = (float*)carve(sizeof(float) * (size_t)NM * D);
    float*          hW    = (float*)carve(sizeof(float) * (size_t)NM * D);
    float*          outac = (float*)carve(sizeof(float) * (size_t)NM * D);
    unsigned short* hb    = (unsigned short*)carve(sizeof(short) * (size_t)NM * D);
    unsigned short* Wt    = (unsigned short*)carve(sizeof(short) * D * D);
    float*          ssrc  = (float*)carve(sizeof(float) * NM);
    float*          sdst  = (float*)carve(sizeof(float) * NM);
    float*          logits= (float*)carve(sizeof(float) * NEDGE);
    float*          exv   = (float*)carve(sizeof(float) * NEDGE);
    unsigned*       mkey  = (unsigned*)carve(sizeof(unsigned) * NM);
    float*          denom = (float*)carve(sizeof(float) * NM);
    float*          ce    = (float*)carve(256);
    float*          hw5   = (float*)carve(sizeof(float) * NM);
    float*          outf  = (float*)carve(sizeof(float) * NM);

    const long ND = (long)NM * D;   // 8,192,000

    k_input_proj<<<NM, D, 0, stream>>>(x, pW, pb, cvid, vge, vnd, h, hb);

    for (int l = 0; l < 4; ++l) {
        const float* Wl  = Ws + (size_t)l * D * D;
        const float* bl  = bs + (size_t)l * D;
        const float* asl = a_srcs + (size_t)l * D;
        const float* adl = a_dsts + (size_t)l * D;
        const float* ael = a_edges + (size_t)l * D;
        const float* Wel = Wes + (size_t)l * EDGE_DIM * D;

        k_w_transpose<<<(D * D) / 256, 256, 0, stream>>>(Wl, Wt);
        k_ce<<<1, 32, 0, stream>>>(Wel, ael, ce);
        k_gemm_bf16<<<NM / 64, 256, 0, stream>>>(hb, Wt, hW);
        k_node_scores<<<NM / 8, 256, 0, stream>>>(hW, asl, adl, ssrc, sdst);
        k_zero<<<(int)(ND / 256), 256, 0, stream>>>((unsigned*)outac, ND);
        k_zero<<<(NM + 255) / 256, 256, 0, stream>>>(mkey, NM);
        k_zero<<<(NM + 255) / 256, 256, 0, stream>>>((unsigned*)denom, NM);
        k_edge_logits<<<NEDGE / 256, 256, 0, stream>>>(ei, ea, ssrc, sdst, ce, logits, mkey);
        k_edge_exp<<<NEDGE / 256, 256, 0, stream>>>(ei, logits, mkey, exv, denom);
        k_edge_aggregate<<<NEDGE / 8, 256, 0, stream>>>(ei, ea, hW, Wel, exv, denom, outac);
        k_bias_act<<<(int)(ND / 256), 256, 0, stream>>>(outac, bl, h, hb, ND);
    }

    // output layer (128 -> 1), no activation
    k_final_node<<<NM / 8, 256, 0, stream>>>(h, W5, hw5);
    k_zero<<<(NM + 255) / 256, 256, 0, stream>>>(mkey, NM);
    k_zero<<<(NM + 255) / 256, 256, 0, stream>>>((unsigned*)denom, NM);
    k_zero<<<(NM + 255) / 256, 256, 0, stream>>>((unsigned*)outf, NM);
    k_edge_logits5<<<NEDGE / 256, 256, 0, stream>>>(ei, ea, hw5, a_src5, a_dst5, a_edge5,
                                                    We5, logits, mkey);
    k_edge_exp<<<NEDGE / 256, 256, 0, stream>>>(ei, logits, mkey, exv, denom);
    k_edge_agg5<<<NEDGE / 256, 256, 0, stream>>>(ei, ea, hw5, We5, exv, denom, outf);
    k_final_out<<<NM / 256, 256, 0, stream>>>(outf, b5, y);
}
